// WEGATModule_70695161692581
// MI455X (gfx1250) — compile-verified, matrix operand order checked by reading, not run on hardware
//
#include <hip/hip_runtime.h>
#include <hip/hip_bf16.h>
#include <math.h>

// ---------------------------------------------------------------------------
// WEGAT (3x GAT-with-edge-features + TopK pool) + MLP head, for gfx1250.
// Node-feature GEMMs: v_wmma_f32_16x16x32_f16, weights staged to LDS with
// global_load_async_to_lds_b128 (ASYNCcnt) and read back as WMMA B-fragments.
// ---------------------------------------------------------------------------

typedef __attribute__((ext_vector_type(16))) _Float16 v16h;
typedef __attribute__((ext_vector_type(8)))  _Float16 v8h;
typedef __attribute__((ext_vector_type(8)))  float    v8f;

#define HID 64
#define HEADS 4
#define NOUT 256           // HEADS*HID
#define NUMEDGE 3
#define NUMCHIP 18
#define NB 8               // batch (graphs)
#define EDGES 524288

static inline int ceilDiv(long a, long b) { return (int)((a + b - 1) / b); }

// ---------------- staging: fp32 -> f16, zero-padded K ----------------------
__global__ void stage_a_f16(const float* __restrict__ src, _Float16* __restrict__ dst,
                            int M, int Kin, int Kp) {
  long i = (long)blockIdx.x * blockDim.x + threadIdx.x;
  if (i >= (long)M * Kp) return;
  int k = (int)(i % Kp);
  long m = i / Kp;
  dst[i] = (k < Kin) ? (_Float16)src[m * Kin + k] : (_Float16)0.0f;
}

// W [Kin, 256] fp32 -> Wt [256, Kp] f16 (transposed, padded)
__global__ void stage_wt_f16(const float* __restrict__ W, _Float16* __restrict__ Wt,
                             int Kin, int Kp) {
  long i = (long)blockIdx.x * blockDim.x + threadIdx.x;
  if (i >= (long)NOUT * Kp) return;
  int k = (int)(i % Kp);
  int n = (int)(i / Kp);
  Wt[i] = (k < Kin) ? (_Float16)W[(long)k * NOUT + n] : (_Float16)0.0f;
}

// ---------------- WMMA GEMM: C[M,256] = A[M,Kp] * Wt^T + bias --------------
// Block = 4 waves = 64 rows x one 16-col n-tile. The 16xKP weight tile is
// async-copied to LDS once per block (same IOFFSET applies to LDS and global
// addresses), then each wave computes one 16x16 tile with KP/32 WMMAs.
template <int KP>
__global__ __launch_bounds__(128)
void gemm_wmma_lds(const _Float16* __restrict__ A, const _Float16* __restrict__ Bt,
                   const float* __restrict__ bias, float* __restrict__ C, int M) {
  __shared__ _Float16 sB[16 * KP];
  const int ntile = blockIdx.x & 15;
  const int mbase = (blockIdx.x >> 4) << 6;     // 64 rows per block
  const int wave  = threadIdx.x >> 5;
  const int lane  = threadIdx.x & 31;
  const int colL  = lane & 15;
  const int kb    = (lane >> 4) << 3;           // 0 or 8

  if (wave == 0) {
    // per-lane LDS byte address (low 32 bits of the flat pointer == LDS offset)
    unsigned lb = (unsigned)(uintptr_t)(void*)sB + (unsigned)lane * 16u;
    const char* gp = (const char*)(Bt + (size_t)ntile * 16 * KP) + lane * 16;
    asm volatile("global_load_async_to_lds_b128 %0, %1, off"
                 :: "v"(lb), "v"(gp) : "memory");
    asm volatile("global_load_async_to_lds_b128 %0, %1, off offset:512"
                 :: "v"(lb), "v"(gp) : "memory");
    if (KP == 64) {
      asm volatile("global_load_async_to_lds_b128 %0, %1, off offset:1024"
                   :: "v"(lb), "v"(gp) : "memory");
      asm volatile("global_load_async_to_lds_b128 %0, %1, off offset:1536"
                   :: "v"(lb), "v"(gp) : "memory");
    }
    asm volatile("s_wait_asynccnt 0x0" ::: "memory");
  }
  __syncthreads();

  const int row = mbase + wave * 16 + colL;
  v8f acc = {};
#pragma unroll
  for (int kk = 0; kk < KP; kk += 32) {
    const v8h* pa = (const v8h*)(A + (size_t)row * KP + kk + kb);
    v8h a0 = pa[0], a1 = pa[2];                 // K = kb..kb+7, kb+16..kb+23
    const v8h* pb = (const v8h*)(&sB[colL * KP + kk + kb]);
    v8h b0 = pb[0], b1 = pb[2];
    v16h av, bv;
#pragma unroll
    for (int i = 0; i < 8; ++i) {
      av[i] = a0[i]; av[8 + i] = a1[i];
      bv[i] = b0[i]; bv[8 + i] = b1[i];
    }
    acc = __builtin_amdgcn_wmma_f32_16x16x32_f16(
        false, av, false, bv, (short)0, acc, false, false);
  }
  const int col   = ntile * 16 + colL;
  const float bb  = bias[col];
  const int rbase = mbase + wave * 16 + ((lane >> 4) << 3);
#pragma unroll
  for (int r = 0; r < 8; ++r)
    C[(long)(rbase + r) * NOUT + col] = acc[r] + bb;
}

// ---------------- attention scalars ----------------------------------------
__global__ void attn_node(const float* __restrict__ xh,
                          const float* __restrict__ att_src,
                          const float* __restrict__ att_dst,
                          float* __restrict__ a_s, float* __restrict__ a_d, int N) {
  long i = (long)blockIdx.x * blockDim.x + threadIdx.x;
  if (i >= (long)N * HEADS) return;
  long n = i >> 2; int h = (int)(i & 3);
  const float* v = xh + n * NOUT + h * HID;
  float ss = 0.f, sd = 0.f;
#pragma unroll 8
  for (int c = 0; c < HID; ++c) {
    float x = v[c];
    ss += x * att_src[h * HID + c];
    sd += x * att_dst[h * HID + c];
  }
  a_s[i] = ss; a_d[i] = sd;
}

// edge linear + a_e + mean-over-heads (next-layer e), in-place safe per-edge
__global__ void edge_feat(const float* __restrict__ e_in,
                          const float* __restrict__ We, const float* __restrict__ be,
                          const float* __restrict__ att_e,
                          float* __restrict__ e_out, float* __restrict__ a_e, long E) {
  long e = (long)blockIdx.x * blockDim.x + threadIdx.x;
  if (e >= E) return;
  float x0 = e_in[e * 3 + 0], x1 = e_in[e * 3 + 1], x2 = e_in[e * 3 + 2];
  float mean[3] = {0.f, 0.f, 0.f};
#pragma unroll
  for (int h = 0; h < HEADS; ++h) {
    float ae = 0.f;
#pragma unroll
    for (int c = 0; c < 3; ++c) {
      int col = h * 3 + c;
      float v = x0 * We[0 * 12 + col] + x1 * We[1 * 12 + col] + x2 * We[2 * 12 + col] + be[col];
      ae += v * att_e[col];
      mean[c] += v;
    }
    a_e[e * 4 + h] = ae;
  }
#pragma unroll
  for (int c = 0; c < 3; ++c) e_out[e * 3 + c] = fmaxf(mean[c] * 0.25f, 0.f);
}

// ---------------- edge softmax (segment max / sum via atomics) -------------
__device__ inline void atomicMaxF(float* addr, float val) {
  if (val >= 0.f) atomicMax((int*)addr, __float_as_int(val));
  else            atomicMin((unsigned int*)addr, (unsigned int)__float_as_int(val));
}

__global__ void fill_f32(float* __restrict__ p, float v, long n) {
  long i = (long)blockIdx.x * blockDim.x + threadIdx.x;
  if (i < n) p[i] = v;
}

__global__ void edge_logits(const int* __restrict__ src, const int* __restrict__ dst,
                            const unsigned char* __restrict__ mask,
                            const float* __restrict__ a_s, const float* __restrict__ a_d,
                            const float* __restrict__ a_e,
                            float* __restrict__ lw, float* __restrict__ m, long E) {
  long i = (long)blockIdx.x * blockDim.x + threadIdx.x;
  if (i >= E * 4) return;
  long e = i >> 2; int h = (int)(i & 3);
  if (!mask[e]) { lw[i] = 0.f; return; }
  float l = a_s[(long)src[e] * 4 + h] + a_d[(long)dst[e] * 4 + h] + a_e[i];
  l = (l > 0.f) ? l : 0.2f * l;     // leaky_relu
  lw[i] = l;
  atomicMaxF(&m[(long)dst[e] * 4 + h], l);
}

__global__ void edge_exp(const int* __restrict__ dst, const unsigned char* __restrict__ mask,
                         const float* __restrict__ m, float* __restrict__ lw,
                         float* __restrict__ denom, long E) {
  long i = (long)blockIdx.x * blockDim.x + threadIdx.x;
  if (i >= E * 4) return;
  long e = i >> 2; int h = (int)(i & 3);
  if (!mask[e]) { lw[i] = 0.f; return; }
  float mm = m[(long)dst[e] * 4 + h];
  if (!isfinite(mm)) mm = 0.f;
  float w = expf(lw[i] - mm);
  lw[i] = w;
  atomicAdd(&denom[(long)dst[e] * 4 + h], w);
}

__global__ void edge_agg(const int* __restrict__ src, const int* __restrict__ dst,
                         const unsigned char* __restrict__ mask,
                         const float* __restrict__ lw, const float* __restrict__ denom,
                         const float* __restrict__ xh, float* __restrict__ out, long E) {
  long i = (long)blockIdx.x * blockDim.x + threadIdx.x;
  if (i >= E * 4) return;
  long e = i >> 2; int h = (int)(i & 3);
  if (!mask[e]) return;
  float d = denom[(long)dst[e] * 4 + h];
  float alpha = lw[i] / fmaxf(d, 1e-16f);
  if (alpha == 0.f) return;
  const float* xs = xh  + (long)src[e] * NOUT + h * HID;
  float*       o  = out + (long)dst[e] * NOUT + h * HID;
#pragma unroll 4
  for (int c = 0; c < HID; ++c) atomicAdd(&o[c], alpha * xs[c]);
}

__global__ void node_finalize(const float* __restrict__ agg, float* __restrict__ xo, int N) {
  long i = (long)blockIdx.x * blockDim.x + threadIdx.x;
  if (i >= (long)N * HID) return;
  long n = i >> 6; int c = (int)(i & 63);
  const float* a = agg + n * NOUT;
  float v = (a[c] + a[HID + c] + a[2 * HID + c] + a[3 * HID + c]) * 0.25f;
  xo[i] = fmaxf(v, 0.f);
}

// ---------------- TopK pooling (ratio 0.5) ---------------------------------
__global__ void pool_score(const float* __restrict__ x, const float* __restrict__ w,
                           float* __restrict__ score, int N) {
  long n = (long)blockIdx.x * blockDim.x + threadIdx.x;
  if (n >= N) return;
  float nw = 0.f, s = 0.f;
#pragma unroll 8
  for (int c = 0; c < HID; ++c) { nw += w[c] * w[c]; s += x[n * HID + c] * w[c]; }
  score[n] = s * rsqrtf(nw);
}

// one block per graph; stable descending rank reproduces lax.top_k ordering
__global__ void pool_rank(const float* __restrict__ score, int n, int k,
                          int* __restrict__ old2new, int* __restrict__ new2old,
                          float* __restrict__ tanhs) {
  extern __shared__ float sh[];
  int b = blockIdx.x;
  const float* s = score + (long)b * n;
  for (int i = threadIdx.x; i < n; i += blockDim.x) sh[i] = s[i];
  __syncthreads();
  for (int i = threadIdx.x; i < n; i += blockDim.x) {
    float si = sh[i];
    int cnt = 0;
    for (int j = 0; j < n; ++j) {
      float sj = sh[j];
      cnt += (sj > si) || (sj == si && j < i);
    }
    if (cnt < k) {
      int nid = b * k + cnt;
      old2new[(long)b * n + i] = nid;
      new2old[nid]   = b * n + i;
      tanhs[nid]     = tanhf(si);
    } else {
      old2new[(long)b * n + i] = -1;
    }
  }
}

__global__ void pool_gather(const float* __restrict__ x, const int* __restrict__ new2old,
                            const float* __restrict__ tanhs, float* __restrict__ xo, int Nn) {
  long i = (long)blockIdx.x * blockDim.x + threadIdx.x;
  if (i >= (long)Nn * HID) return;
  long m = i >> 6; int c = (int)(i & 63);
  xo[i] = x[(long)new2old[m] * HID + c] * tanhs[m];
}

__global__ void edge_remap(const int* __restrict__ src, const int* __restrict__ dst,
                           const unsigned char* __restrict__ mask,
                           const int* __restrict__ o2n,
                           int* __restrict__ nsrc, int* __restrict__ ndst,
                           unsigned char* __restrict__ nmask, long E) {
  long e = (long)blockIdx.x * blockDim.x + threadIdx.x;
  if (e >= E) return;
  int ok = 0, s = 0, d = 0;
  if (mask[e]) {
    int a = o2n[src[e]], b = o2n[dst[e]];
    if (a >= 0 && b >= 0) { ok = 1; s = a; d = b; }
  }
  nsrc[e] = s; ndst[e] = d; nmask[e] = (unsigned char)ok;
}

// ---------------- head: global max pool + MLPs -----------------------------
__global__ void global_max(const float* __restrict__ x, float* __restrict__ g, int nf) {
  int i = blockIdx.x * blockDim.x + threadIdx.x;
  if (i >= NB * HID) return;
  int b = i >> 6, c = i & 63;
  float m = -INFINITY;
  for (int j = 0; j < nf; ++j) m = fmaxf(m, x[((long)b * nf + j) * HID + c]);
  g[i] = m;
}

struct HeadParams {
  const float* lw[5]; const float* lb[5];
  const float* pw[5]; const float* pb[5];
  const float* rw; const float* rb;
};

__global__ void head_mlp(const float* __restrict__ g, const float* __restrict__ prom,
                         HeadParams P, float* __restrict__ out) {
  int b = threadIdx.x;
  if (b >= NB) return;
  const int dims[6]  = {64, 15, 15, 10, 5, 2};
  const int pdims[6] = {18, 15, 15, 10, 5, 2};
  float buf[64], tmp[15];
  for (int c = 0; c < 64; ++c) buf[c] = g[b * 64 + c];
  for (int l = 0; l < 5; ++l) {
    int fi = dims[l], fo = dims[l + 1];
    for (int o = 0; o < fo; ++o) {
      float s = P.lb[l][o];
      for (int i = 0; i < fi; ++i) s += buf[i] * P.lw[l][i * fo + o];
      tmp[o] = fmaxf(s, 0.f);
    }
    for (int o = 0; o < fo; ++o) buf[o] = tmp[o];
  }
  float pbuf[18];
  for (int c = 0; c < 18; ++c) pbuf[c] = prom[b * 18 + c];
  for (int l = 0; l < 5; ++l) {
    int fi = pdims[l], fo = pdims[l + 1];
    for (int o = 0; o < fo; ++o) {
      float s = P.pb[l][o];
      for (int i = 0; i < fi; ++i) s += pbuf[i] * P.pw[l][i * fo + o];
      tmp[o] = fmaxf(s, 0.f);
    }
    for (int o = 0; o < fo; ++o) pbuf[o] = tmp[o];
  }
  out[b] = P.rb[0] + buf[0] * P.rw[0] + buf[1] * P.rw[1]
                   + pbuf[0] * P.rw[2] + pbuf[1] * P.rw[3];
}

// ---------------------------------------------------------------------------
extern "C" void kernel_launch(void* const* d_in, const int* in_sizes, int n_in,
                              void* d_out, int out_size, void* d_ws, size_t ws_size,
                              hipStream_t stream) {
  const long E = EDGES;
  // ---- input index mapping: detect pytree flatten convention --------------
  // insertion order: x, edge_attr, prom_x, edge_index, params(...)
  // sorted-key order: edge_attr, edge_index, params(...), prom_x, x
  bool sorted = (in_sizes[0] == (int)(E * 3));
  int iX, iEA, iPROM, iEI, iRoW, iRoB, linB, promB;
  int cWn[3], cbn[3], cWe[3], cbe[3], cas_[3], cad[3], cae[3], cpw[3];
  if (!sorted) {
    iX = 0; iEA = 1; iPROM = 2; iEI = 3;
    for (int l = 0; l < 3; ++l) {
      int b = 4 + 8 * l;
      cWn[l] = b; cbn[l] = b + 1; cWe[l] = b + 2; cbe[l] = b + 3;
      cas_[l] = b + 4; cad[l] = b + 5; cae[l] = b + 6; cpw[l] = b + 7;
    }
    linB = 28; promB = 38; iRoW = 48; iRoB = 49;
  } else {
    iEA = 0; iEI = 1;
    for (int l = 0; l < 3; ++l) {
      int b = 2 + 8 * l;   // sorted keys: We, Wn, att_dst, att_edge, att_src, be, bn, pool_w
      cWe[l] = b; cWn[l] = b + 1; cad[l] = b + 2; cae[l] = b + 3;
      cas_[l] = b + 4; cbe[l] = b + 5; cbn[l] = b + 6; cpw[l] = b + 7;
    }
    linB = 26; promB = 36; iRoW = 46; iRoB = 47; iPROM = 48; iX = 49;
  }
  const float* X    = (const float*)d_in[iX];
  const float* EA   = (const float*)d_in[iEA];
  const float* PROM = (const float*)d_in[iPROM];
  const int*   EI   = (const int*)d_in[iEI];

  // ---- workspace carving --------------------------------------------------
  char* base = (char*)d_ws; size_t off = 0;
  auto alloc = [&](size_t bytes) -> void* {
    void* p = base + off; off = (off + bytes + 255) & ~(size_t)255; return p;
  };
  const int N0 = 32768;
  _Float16* xh16 = (_Float16*)alloc((size_t)N0 * 64 * 2);
  _Float16* wt16 = (_Float16*)alloc((size_t)NOUT * 64 * 2);
  float* xh    = (float*)alloc((size_t)N0 * NOUT * 4);
  float* agg   = (float*)alloc((size_t)N0 * NOUT * 4);
  float* a_s   = (float*)alloc((size_t)N0 * 4 * 4);
  float* a_d   = (float*)alloc((size_t)N0 * 4 * 4);
  float* a_e   = (float*)alloc((size_t)E * 4 * 4);
  float* lw    = (float*)alloc((size_t)E * 4 * 4);
  float* e_buf = (float*)alloc((size_t)E * 3 * 4);
  float* mmax  = (float*)alloc((size_t)N0 * 4 * 4);
  float* denom = (float*)alloc((size_t)N0 * 4 * 4);
  float* xfin  = (float*)alloc((size_t)N0 * 64 * 4);
  float* xpA   = (float*)alloc((size_t)(N0 / 2) * 64 * 4);
  float* xpB   = (float*)alloc((size_t)(N0 / 2) * 64 * 4);
  float* score = (float*)alloc((size_t)N0 * 4);
  int*   o2n   = (int*)alloc((size_t)N0 * 4);
  int*   n2o   = (int*)alloc((size_t)(N0 / 2) * 4);
  float* tanhs = (float*)alloc((size_t)(N0 / 2) * 4);
  int*   srcA  = (int*)alloc((size_t)E * 4);
  int*   dstA  = (int*)alloc((size_t)E * 4);
  int*   srcB  = (int*)alloc((size_t)E * 4);
  int*   dstB  = (int*)alloc((size_t)E * 4);
  unsigned char* mskA = (unsigned char*)alloc((size_t)E);
  unsigned char* mskB = (unsigned char*)alloc((size_t)E);
  float* g = (float*)alloc((size_t)NB * HID * 4);
  (void)ws_size;

  // ---- init edge buffers --------------------------------------------------
  hipMemcpyAsync(srcA, EI,     E * sizeof(int), hipMemcpyDeviceToDevice, stream);
  hipMemcpyAsync(dstA, EI + E, E * sizeof(int), hipMemcpyDeviceToDevice, stream);
  hipMemsetAsync(mskA, 1, E, stream);

  int* scur = srcA; int* dcur = dstA; unsigned char* mcur = mskA;
  int* snxt = srcB; int* dnxt = dstB; unsigned char* mnxt = mskB;
  const float* xcur = X;

  const int Ns[3]  = {32768, 16384, 8192};
  const int fin[3] = {NUMCHIP, HID, HID};
  const int Kps[3] = {32, 64, 64};
  const int TPB = 256;

  for (int l = 0; l < 3; ++l) {
    int N = Ns[l], Kp = Kps[l];
    // stage A + Wt, run WMMA GEMM (B tile via async LDS copy)
    stage_a_f16<<<ceilDiv((long)N * Kp, TPB), TPB, 0, stream>>>(
        (l == 0) ? X : xcur, xh16, N, fin[l], Kp);
    stage_wt_f16<<<ceilDiv((long)NOUT * Kp, TPB), TPB, 0, stream>>>(
        (const float*)d_in[cWn[l]], wt16, fin[l], Kp);
    if (Kp == 32)
      gemm_wmma_lds<32><<<(N / 64) * 16, 128, 0, stream>>>(
          xh16, wt16, (const float*)d_in[cbn[l]], xh, N);
    else
      gemm_wmma_lds<64><<<(N / 64) * 16, 128, 0, stream>>>(
          xh16, wt16, (const float*)d_in[cbn[l]], xh, N);

    attn_node<<<ceilDiv((long)N * 4, TPB), TPB, 0, stream>>>(
        xh, (const float*)d_in[cas_[l]], (const float*)d_in[cad[l]], a_s, a_d, N);
    edge_feat<<<ceilDiv(E, TPB), TPB, 0, stream>>>(
        (l == 0) ? EA : e_buf, (const float*)d_in[cWe[l]],
        (const float*)d_in[cbe[l]], (const float*)d_in[cae[l]], e_buf, a_e, E);

    fill_f32<<<ceilDiv((long)N * 4, TPB), TPB, 0, stream>>>(mmax, -INFINITY, (long)N * 4);
    hipMemsetAsync(denom, 0, (size_t)N * 4 * sizeof(float), stream);
    hipMemsetAsync(agg,   0, (size_t)N * NOUT * sizeof(float), stream);

    edge_logits<<<ceilDiv(E * 4, TPB), TPB, 0, stream>>>(
        scur, dcur, mcur, a_s, a_d, a_e, lw, mmax, E);
    edge_exp<<<ceilDiv(E * 4, TPB), TPB, 0, stream>>>(dcur, mcur, mmax, lw, denom, E);
    edge_agg<<<ceilDiv(E * 4, TPB), TPB, 0, stream>>>(scur, dcur, mcur, lw, denom, xh, agg, E);
    node_finalize<<<ceilDiv((long)N * HID, TPB), TPB, 0, stream>>>(agg, xfin, N);

    // TopK pool (ratio 0.5)
    int n = N / NB, k = n / 2;
    pool_score<<<ceilDiv((long)N, TPB), TPB, 0, stream>>>(
        xfin, (const float*)d_in[cpw[l]], score, N);
    pool_rank<<<NB, 256, n * sizeof(float), stream>>>(score, n, k, o2n, n2o, tanhs);
    float* xpool = (l & 1) ? xpB : xpA;
    pool_gather<<<ceilDiv((long)(N / 2) * HID, TPB), TPB, 0, stream>>>(
        xfin, n2o, tanhs, xpool, N / 2);
    edge_remap<<<ceilDiv(E, TPB), TPB, 0, stream>>>(
        scur, dcur, mcur, o2n, snxt, dnxt, mnxt, E);

    { int* t; unsigned char* tm;
      t = scur; scur = snxt; snxt = t;
      t = dcur; dcur = dnxt; dnxt = t;
      tm = mcur; mcur = mnxt; mnxt = tm; }
    xcur = xpool;
  }

  // head: global max pool over 512 nodes/graph, then MLPs + readout
  global_max<<<ceilDiv((long)NB * HID, TPB), TPB, 0, stream>>>(xcur, g, 512);
  HeadParams P;
  for (int i = 0; i < 5; ++i) {
    P.lw[i] = (const float*)d_in[linB  + 2 * i]; P.lb[i] = (const float*)d_in[linB  + 2 * i + 1];
    P.pw[i] = (const float*)d_in[promB + 2 * i]; P.pb[i] = (const float*)d_in[promB + 2 * i + 1];
  }
  P.rw = (const float*)d_in[iRoW]; P.rb = (const float*)d_in[iRoB];
  head_mlp<<<1, 32, 0, stream>>>(g, PROM, P, (float*)d_out);
  (void)n_in; (void)out_size;
}
